// MultiRelationGraphAttention_79018808312383
// MI455X (gfx1250) — matrix operand order repaired and checked by reading, not compiled, not run on hardware
//
#include <hip/hip_runtime.h>
#include <hip/hip_bf16.h>

#define BB 8
#define NN 2048
#define DD 256
#define FF 128
#define RR 2

typedef __attribute__((ext_vector_type(16))) __bf16 v16bf;
typedef __attribute__((ext_vector_type(8)))  __bf16 v8bf;
typedef __attribute__((ext_vector_type(8)))  float  v8f;

union V16 { v16bf v; v8bf h[2]; };

// native bf16 convert (gfx1250 has HW bf16 cvt; fall back is inline-expanded)
static __device__ __forceinline__ __bf16 f2bf(float x) { return (__bf16)x; }

// ---------------- workspace layout (bytes) ----------------
static constexpr size_t O_X1   = 0;                                   // B*F f32
static constexpr size_t O_C    = 4096;                                // 5*B*D f32
static constexpr size_t O_WT   = 45056;                               // 5*256*256 bf16 (transposed)
static constexpr size_t O_WFT  = O_WT  + (size_t)5*256*256*2;         // 256*512 bf16 (transposed)
static constexpr size_t O_Q    = O_WFT + (size_t)512*256*2;           // B*N*D bf16
static constexpr size_t O_K    = O_Q   + (size_t)BB*NN*DD*2;          // B*R*N*D bf16
static constexpr size_t O_VT   = O_K   + (size_t)BB*RR*NN*DD*2;       // B*R*D*N bf16 (transposed)
static constexpr size_t O_M    = O_VT  + (size_t)BB*RR*NN*DD*2;       // B*N*(R*D) bf16
static constexpr size_t O_NE   = O_M   + (size_t)BB*NN*RR*DD*2;       // B*N*D bf16 copy of node_embed
static constexpr size_t O_MASK = O_NE  + (size_t)BB*NN*DD*2;          // B*R*N*(N/32) u32 bitmask

// ---------------- prep kernels ----------------
__global__ void prep_x1(const float* __restrict__ inx, const float* __restrict__ wqi_w,
                        const float* __restrict__ wqi_b, float* __restrict__ x1) {
  int b = blockIdx.x, f = threadIdx.x;
  float s = wqi_b[f];
  for (int j = 0; j < FF; ++j) s += inx[b*FF + j] * wqi_w[j*FF + f];
  x1[b*FF + f] = s;
}

__global__ void prep_consts(const float* __restrict__ x1,
                            const float* __restrict__ wq_w, const float* __restrict__ wq_b,
                            const float* __restrict__ wk_w, const float* __restrict__ wk_b,
                            const float* __restrict__ wv_w, const float* __restrict__ wv_b,
                            float* __restrict__ cvec) {
  int mat = blockIdx.x % 5, b = blockIdx.x / 5, d = threadIdx.x;
  const float* W; const float* bias;
  if (mat == 0)      { W = wq_w;               bias = wq_b; }
  else if (mat == 1) { W = wk_w;               bias = wk_b; }
  else if (mat == 2) { W = wk_w + 384*256;     bias = wk_b + 256; }
  else if (mat == 3) { W = wv_w;               bias = wv_b; }
  else               { W = wv_w + 384*256;     bias = wv_b + 256; }
  float s = bias[d];
  for (int f = 0; f < FF; ++f) s += x1[b*FF + f] * W[(256 + f)*256 + d];
  cvec[((size_t)mat*BB + b)*DD + d] = s;
}

__global__ void prep_wt(const float* __restrict__ wq_w, const float* __restrict__ wk_w,
                        const float* __restrict__ wv_w, const float* __restrict__ wf_w,
                        __bf16* __restrict__ wt, __bf16* __restrict__ wft) {
  int idx = blockIdx.x * blockDim.x + threadIdx.x;
  if (idx < 5*65536) {
    int mat = idx >> 16, rem = idx & 65535, dcol = rem >> 8, dp = rem & 255;
    const float* W = (mat == 0) ? wq_w
                   : (mat <= 2) ? wk_w + (size_t)(mat-1)*384*256
                                : wv_w + (size_t)(mat-3)*384*256;
    wt[idx] = f2bf(W[dp*256 + dcol]);             // wt[mat][dcol][dp] = W[dp][dcol]
  } else if (idx < 5*65536 + 512*256) {
    int j = idx - 5*65536, dcol = j >> 9, k = j & 511;
    wft[(size_t)dcol*512 + k] = f2bf(wf_w[k*256 + dcol]);
  }
}

// node_embed fp32 -> bf16 copy (done once; removes 80x re-convert in proj)
__global__ void prep_ne(const float* __restrict__ ne, __bf16* __restrict__ nebf) {
  size_t i = (size_t)blockIdx.x * blockDim.x + threadIdx.x;
  nebf[i] = f2bf(ne[i]);
}

// adj int32 -> 1-bit mask, streaming non-temporal read + wave32 ballot
__global__ void pack_adj(const int* __restrict__ adj, unsigned* __restrict__ mask) {
  size_t idx = (size_t)blockIdx.x * blockDim.x + threadIdx.x;
  int a = __builtin_nontemporal_load(adj + idx);
  unsigned long long bits = __ballot(a != 0);     // wave32: low 32 bits valid
  if ((threadIdx.x & 31) == 0) mask[idx >> 5] = (unsigned)bits;
}

// ---------------- projection GEMM: q, k(r), v(r) ----------------
__global__ void __launch_bounds__(256) proj_kernel(
    const __bf16* __restrict__ nebf, const float* __restrict__ cvec,
    const __bf16* __restrict__ wt, __bf16* __restrict__ qout,
    __bf16* __restrict__ kout, __bf16* __restrict__ vtout) {
  int wave = blockIdx.x * (blockDim.x >> 5) + (threadIdx.x >> 5);
  int lane = threadIdx.x & 31, l16 = lane & 15, half = lane >> 4;
  int dt = wave & 15, nt = (wave >> 4) & 127, mat = (wave >> 11) % 5, b = wave / 10240;
  int rowbase = nt * 16, dcol = dt * 16 + l16;
  int klo = half * 8, kb16 = half * 16;

  const __bf16* arow = nebf + ((size_t)b*NN + rowbase + l16) * DD;
  const __bf16* wrow = wt + ((size_t)mat*DD + dcol) * DD;
  v8f acc = {0.f,0.f,0.f,0.f,0.f,0.f,0.f,0.f};
#pragma unroll
  for (int kc = 0; kc < 8; ++kc) {
    V16 a;
    a.h[0] = *(const v8bf*)(arow + kc*32 + kb16);
    a.h[1] = *(const v8bf*)(arow + kc*32 + kb16 + 8);
    v16bf bf = *(const v16bf*)(wrow + kc*32 + kb16);
    acc = __builtin_amdgcn_wmma_f32_16x16x32_bf16(false, a.v, false, bf, (short)0, acc, false, false);
  }
  float cadd = cvec[((size_t)mat*BB + b)*DD + dcol];
#pragma unroll
  for (int i = 0; i < 8; ++i) acc[i] += cadd;

  if (mat == 0) {
    __bf16* outp = qout + (size_t)b*NN*DD;
#pragma unroll
    for (int i = 0; i < 8; ++i)
      outp[(size_t)(rowbase + half*8 + i)*DD + dcol] = f2bf(acc[i]);
  } else if (mat <= 2) {
    __bf16* outp = kout + (size_t)(b*RR + (mat-1))*NN*DD;
#pragma unroll
    for (int i = 0; i < 8; ++i)
      outp[(size_t)(rowbase + half*8 + i)*DD + dcol] = f2bf(acc[i]);
  } else {
    __bf16* outp = vtout + ((size_t)(b*RR + (mat-3))*DD + dcol)*NN + rowbase + half*8;
    v8bf pk;
#pragma unroll
    for (int i = 0; i < 8; ++i) pk[i] = f2bf(acc[i]);
    *(v8bf*)outp = pk;
  }
}

// ---------------- flash attention per (b, r, 16-row tile) ----------------
__global__ void __launch_bounds__(256, 1) attn_kernel(
    const __bf16* __restrict__ qbf, const __bf16* __restrict__ kbf,
    const __bf16* __restrict__ vtbf, const unsigned* __restrict__ mask,
    __bf16* __restrict__ mout) {
  __shared__ __bf16 plds[8 * 16 * 32];
  int widb = threadIdx.x >> 5;
  int wave = blockIdx.x * 8 + widb;
  int lane = threadIdx.x & 31, l16 = lane & 15, half = lane >> 4;
  int rt = wave & 127, r = (wave >> 7) & 1, b = wave >> 8;
  int rowbase = rt * 16, klo = half * 8, kb16 = half * 16;

  V16 qa[8];
  const __bf16* qrow = qbf + ((size_t)b*NN + rowbase + l16) * DD;
#pragma unroll
  for (int kc = 0; kc < 8; ++kc) {
    qa[kc].h[0] = *(const v8bf*)(qrow + kc*32 + kb16);
    qa[kc].h[1] = *(const v8bf*)(qrow + kc*32 + kb16 + 8);
  }

  v8f o[16];
  const v8f vzero = {0.f,0.f,0.f,0.f,0.f,0.f,0.f,0.f};
#pragma unroll
  for (int t = 0; t < 16; ++t) o[t] = vzero;
  float rowm[8], rowl[8];
#pragma unroll
  for (int i = 0; i < 8; ++i) { rowm[i] = -1e9f; rowl[i] = 0.f; }

  const __bf16*   kbase = kbf  + (size_t)(b*RR + r)*NN*DD;
  const __bf16*   vbase = vtbf + (size_t)(b*RR + r)*DD*NN;
  // this lane prefetches the mask word for row (rowbase + l16); rows are
  // redistributed to owners via intra-16-lane shuffles below
  const unsigned* mrow  = mask + ((size_t)(b*RR + r)*NN + rowbase + l16) * (NN/32);
  __bf16* pl = plds + widb * 512;
  const float scale = 0.0625f;   // 1/sqrt(256)

  for (int ct = 0; ct < 64; ++ct) {
    int colbase = ct * 32;
    v8f s0 = vzero, s1 = vzero;
    const __bf16* krow0 = kbase + (size_t)(colbase + l16) * DD;
    const __bf16* krow1 = kbase + (size_t)(colbase + 16 + l16) * DD;
#pragma unroll
    for (int kc = 0; kc < 8; ++kc) {
      v16bf b0 = *(const v16bf*)(krow0 + kc*32 + kb16);
      s0 = __builtin_amdgcn_wmma_f32_16x16x32_bf16(false, qa[kc].v, false, b0, (short)0, s0, false, false);
      v16bf b1 = *(const v16bf*)(krow1 + kc*32 + kb16);
      s1 = __builtin_amdgcn_wmma_f32_16x16x32_bf16(false, qa[kc].v, false, b1, (short)0, s1, false, false);
    }
    unsigned mw = mrow[ct];                 // 32 mask bits for row (rowbase+l16)
    float p0[8], p1[8], sc[8];
#pragma unroll
    for (int i = 0; i < 8; ++i) {
      unsigned wr = (unsigned)__shfl((int)mw, 8*half + i, 16); // word of my row i
      float v0 = s0[i] * scale, v1 = s1[i] * scale;
      if (!(wr & (1u << l16)))        v0 = -1e9f;
      if (!(wr & (1u << (16 + l16)))) v1 = -1e9f;
      float t = fmaxf(v0, v1);
      t = fmaxf(t, __shfl_xor(t, 1, 16));
      t = fmaxf(t, __shfl_xor(t, 2, 16));
      t = fmaxf(t, __shfl_xor(t, 4, 16));
      t = fmaxf(t, __shfl_xor(t, 8, 16));
      float nm = fmaxf(rowm[i], t);
      sc[i] = __expf(rowm[i] - nm);
      p0[i] = __expf(v0 - nm);
      p1[i] = __expf(v1 - nm);
      float rs = p0[i] + p1[i];
      rs += __shfl_xor(rs, 1, 16);
      rs += __shfl_xor(rs, 2, 16);
      rs += __shfl_xor(rs, 4, 16);
      rs += __shfl_xor(rs, 8, 16);
      rowl[i] = rowl[i]*sc[i] + rs;
      rowm[i] = nm;
    }
#pragma unroll
    for (int t = 0; t < 16; ++t)
#pragma unroll
      for (int i = 0; i < 8; ++i) o[t][i] *= sc[i];
    // stage P (C-layout -> LDS -> A-layout)
#pragma unroll
    for (int i = 0; i < 8; ++i) {
      pl[(half*8 + i)*32 + l16]      = f2bf(p0[i]);
      pl[(half*8 + i)*32 + 16 + l16] = f2bf(p1[i]);
    }
    asm volatile("s_wait_dscnt 0x0" ::: "memory");
    V16 pa;
    pa.h[0] = *(const v8bf*)(pl + l16*32 + kb16);
    pa.h[1] = *(const v8bf*)(pl + l16*32 + kb16 + 8);
#pragma unroll
    for (int t = 0; t < 16; ++t) {
      v16bf bv = *(const v16bf*)(vbase + (size_t)(t*16 + l16)*NN + colbase + kb16);
      o[t] = __builtin_amdgcn_wmma_f32_16x16x32_bf16(false, pa.v, false, bv, (short)0, o[t], false, false);
    }
  }
  float inv[8];
#pragma unroll
  for (int i = 0; i < 8; ++i) inv[i] = 1.0f / rowl[i];
#pragma unroll
  for (int t = 0; t < 16; ++t)
#pragma unroll
    for (int i = 0; i < 8; ++i)
      mout[((size_t)(b*NN + rowbase + half*8 + i))*(RR*DD) + r*DD + t*16 + l16] = f2bf(o[t][i]*inv[i]);
}

// ---------------- fusion GEMM + relu + residual ----------------
__global__ void __launch_bounds__(256) fuse_kernel(
    const __bf16* __restrict__ mcat, const __bf16* __restrict__ wft,
    const float* __restrict__ wfb, const float* __restrict__ ne, float* __restrict__ y) {
  int wave = blockIdx.x * (blockDim.x >> 5) + (threadIdx.x >> 5);
  int lane = threadIdx.x & 31, l16 = lane & 15, half = lane >> 4;
  int dt = wave & 15, nt = (wave >> 4) & 127, b = wave >> 11;
  int rowbase = nt * 16, dcol = dt * 16 + l16;
  int klo = half * 8, kb16 = half * 16;

  const __bf16* arow = mcat + ((size_t)b*NN + rowbase + l16) * (RR*DD);
  const __bf16* brow = wft + (size_t)dcol * (RR*DD);
  v8f acc = {0.f,0.f,0.f,0.f,0.f,0.f,0.f,0.f};
#pragma unroll
  for (int kc = 0; kc < 16; ++kc) {
    V16 a;
    a.h[0] = *(const v8bf*)(arow + kc*32 + kb16);
    a.h[1] = *(const v8bf*)(arow + kc*32 + kb16 + 8);
    v16bf bv = *(const v16bf*)(brow + kc*32 + kb16);
    acc = __builtin_amdgcn_wmma_f32_16x16x32_bf16(false, a.v, false, bv, (short)0, acc, false, false);
  }
  float bias = wfb[dcol];
#pragma unroll
  for (int i = 0; i < 8; ++i) {
    size_t idx = ((size_t)b*NN + rowbase + half*8 + i)*DD + dcol;
    float h = fmaxf(acc[i] + bias, 0.f);
    y[idx] = ne[idx] + h;
  }
}

// ---------------- layernorm (in place on d_out) ----------------
__global__ void __launch_bounds__(256) ln_kernel(float* __restrict__ y,
                                                 const float* __restrict__ g,
                                                 const float* __restrict__ bta) {
  int row = blockIdx.x * 8 + (threadIdx.x >> 5);
  int lane = threadIdx.x & 31;
  float* rp = y + (size_t)row * DD;
  float v[8];
#pragma unroll
  for (int j = 0; j < 8; ++j) v[j] = rp[lane*8 + j];
  float s = 0.f;
#pragma unroll
  for (int j = 0; j < 8; ++j) s += v[j];
  s += __shfl_xor(s, 1, 32);  s += __shfl_xor(s, 2, 32);
  s += __shfl_xor(s, 4, 32);  s += __shfl_xor(s, 8, 32);
  s += __shfl_xor(s, 16, 32);
  float mu = s * (1.0f/256.0f);
  float q = 0.f;
#pragma unroll
  for (int j = 0; j < 8; ++j) { float d = v[j] - mu; q += d*d; }
  q += __shfl_xor(q, 1, 32);  q += __shfl_xor(q, 2, 32);
  q += __shfl_xor(q, 4, 32);  q += __shfl_xor(q, 8, 32);
  q += __shfl_xor(q, 16, 32);
  float rs = rsqrtf(q * (1.0f/256.0f) + 1e-5f);
#pragma unroll
  for (int j = 0; j < 8; ++j)
    rp[lane*8 + j] = (v[j] - mu) * rs * g[lane*8 + j] + bta[lane*8 + j];
}

// ---------------- launcher ----------------
extern "C" void kernel_launch(void* const* d_in, const int* in_sizes, int n_in,
                              void* d_out, int out_size, void* d_ws, size_t ws_size,
                              hipStream_t stream) {
  const float* ne    = (const float*)d_in[0];
  const float* inx   = (const float*)d_in[1];
  const int*   adj   = (const int*)d_in[2];
  const float* wqi_w = (const float*)d_in[3];
  const float* wqi_b = (const float*)d_in[4];
  const float* wq_w  = (const float*)d_in[5];
  const float* wq_b  = (const float*)d_in[6];
  const float* wk_w  = (const float*)d_in[7];
  const float* wk_b  = (const float*)d_in[8];
  const float* wv_w  = (const float*)d_in[9];
  const float* wv_b  = (const float*)d_in[10];
  const float* wf_w  = (const float*)d_in[11];
  const float* wf_b  = (const float*)d_in[12];
  const float* ln_g  = (const float*)d_in[13];
  const float* ln_b  = (const float*)d_in[14];

  char* ws = (char*)d_ws;
  float*    x1   = (float*)(ws + O_X1);
  float*    cvec = (float*)(ws + O_C);
  __bf16*   wt   = (__bf16*)(ws + O_WT);
  __bf16*   wft  = (__bf16*)(ws + O_WFT);
  __bf16*   qb   = (__bf16*)(ws + O_Q);
  __bf16*   kb   = (__bf16*)(ws + O_K);
  __bf16*   vt   = (__bf16*)(ws + O_VT);
  __bf16*   mc   = (__bf16*)(ws + O_M);
  __bf16*   nebf = (__bf16*)(ws + O_NE);
  unsigned* msk  = (unsigned*)(ws + O_MASK);
  float*    y    = (float*)d_out;

  prep_x1<<<BB, FF, 0, stream>>>(inx, wqi_w, wqi_b, x1);
  prep_consts<<<BB*5, DD, 0, stream>>>(x1, wq_w, wq_b, wk_w, wk_b, wv_w, wv_b, cvec);
  prep_wt<<<1792, 256, 0, stream>>>(wq_w, wk_w, wv_w, wf_w, wt, wft);
  prep_ne<<<(BB*NN*DD)/256, 256, 0, stream>>>(ne, nebf);
  pack_adj<<<((size_t)BB*RR*NN*NN)/256, 256, 0, stream>>>(adj, msk);
  proj_kernel<<<10240, 256, 0, stream>>>(nebf, cvec, wt, qb, kb, vt);
  attn_kernel<<<256, 256, 0, stream>>>(qb, kb, vt, msk, mc);
  fuse_kernel<<<2048, 256, 0, stream>>>(mc, wft, wf_b, ne, y);
  ln_kernel<<<2048, 256, 0, stream>>>(y, ln_g, ln_b);
}
